// HybridModel_14353780703250
// MI455X (gfx1250) — matrix operand order repaired and checked by reading, not compiled, not run on hardware
//
#include <hip/hip_runtime.h>
#include <hip/hip_bf16.h>

// ---------------------------------------------------------------------------
// MI455X (gfx1250) implementation notes:
//  * wave32 everywhere; WMMA f32<-f16 16x16x32 for all convs (implicit GEMM)
//  * BatchNorm train-stats: stats pass + recompute pass so the 822MB fp32
//    conv1 activation is NEVER materialized (compute is free vs 23.3 TB/s)
//  * activations are f16 NHWC: pass2 emits ONE global_store_b128 of 8
//    consecutive channels per lane; patch staging is a straight wide copy
//    (global_load_b128 -> ds_store_b128) + global_prefetch of the next
//    row-pair's rows
//  * GEMM K reordered to (r*3+s)*CIN + ci (channels innermost) so A and B
//    fragments are WIDE LDS loads (ds_load_b128); all staging index math is
//    shifts/masks (no div/mod in unrolled code)
//  * conv3 pass2 fuses BN+ReLU+maxpool+global-avg-pool (no h3 tensor at all)
//  * quantum circuit: 5 qubits = 32 amplitudes = one wave32; lane==basis
//    state; gates are shfl_xor butterflies / lane permutes
// ---------------------------------------------------------------------------

typedef _Float16 f16;
typedef __attribute__((ext_vector_type(16))) _Float16 v16h;
typedef __attribute__((ext_vector_type(8)))  _Float16 v8h;
typedef __attribute__((ext_vector_type(8)))  float    v8f;

__device__ __forceinline__ float wred16(float v) {  // reduce within 16-lane half
  v += __shfl_xor(v, 1, 32);
  v += __shfl_xor(v, 2, 32);
  v += __shfl_xor(v, 4, 32);
  v += __shfl_xor(v, 8, 32);
  return v;
}
__device__ __forceinline__ float wred32(float v) {  // full wave32 reduce
  v += __shfl_xor(v, 16, 32);
  return wred16(v);
}

constexpr int clog2(int v) { return v <= 1 ? 0 : 1 + clog2(v / 2); }

// ---------------------------------------------------------------------------
// Fused conv3x3 (SAME) as implicit GEMM via v_wmma_f32_16x16x32_f16.
//  PASS2 == false : conv -> per-channel sum / sumsq atomically into `stats`
//  PASS2 == true  : conv -> BN(scale,shift) -> ReLU -> 2x2 maxpool ->
//                   (FUSE_GAP ? atomic gap accumulation : f16 NHWC b128 store)
// One block = (image b, output row pair yp, 64-column strip). 8 waves.
// Wave owns an (M-tile, col-tile) pair and BOTH rows -> pooling is a single
// shfl_xor(1) exchange, no cross-wave traffic.
//
// Activations (CIN>=16) are NHWC: in[((b*H+y)*W+x)*CIN + ci].
// K ordering: k' = (r*3+s)*CIN + ci.  Weights staged permuted; the patch is
// [row][col][ci] so B fragments are 32B-contiguous -> ds_load_b128 pairs.
// ---------------------------------------------------------------------------
template <typename TIN, int CIN, int COUT, int H, int W, bool PASS2, bool FUSE_GAP>
__global__ __launch_bounds__(256) void conv_wmma(
    const TIN*  __restrict__ in,     // NHWC (CIN==1: NCHW == NHWC)
    const float* __restrict__ wgt,   // [COUT][CIN*9]  (OIHW flattened)
    const float* __restrict__ bias,  // [COUT]
    float*       __restrict__ stats, // [2][COUT]            (pass1)
    const float* __restrict__ scale, // [COUT]               (pass2)
    const float* __restrict__ shift, // [COUT]               (pass2)
    f16*         __restrict__ outp,  // NHWC [B][H/2][W/2][COUT] (pass2,!GAP)
    float*       __restrict__ gap)   // [B][COUT]            (pass2, FUSE_GAP)
{
  constexpr int K      = CIN * 9;
  constexpr int KSTEPS = (K + 31) / 32;
  constexpr int KPAD   = KSTEPS * 32;
  constexpr int MT     = COUT / 16;
  constexpr int NSTRIP = (W + 63) / 64;
  constexpr bool WIDE  = (CIN == 16) || (CIN == 32);
  constexpr int CILOG  = clog2(CIN);

  __shared__ alignas(32) f16   wlds[COUT][KPAD];    // weights, k'-permuted f16
  __shared__ alignas(32) f16   patch[4][68][CIN];   // rows y-1..y+2, ci innermost
  __shared__ float             sred[COUT * 2];      // stats / gap staging

  const int tid  = threadIdx.x;
  const int wave = tid >> 5, lane = tid & 31;
  const int half = lane >> 4, ln = lane & 15;

  int t = blockIdx.x;
  const int strip = t % NSTRIP; t /= NSTRIP;
  const int yp    = t % (H / 2); t /= (H / 2);
  const int b     = t;
  const int xs    = strip * 64;
  const int y0    = 2 * yp - 1;  // global y of patch row 0

  if (tid < COUT * 2) sred[tid] = 0.f;

  // weights -> LDS, permuted to k' = (r*3+s)*CIN + ci  (shift/mask only)
#pragma unroll
  for (int co = wave; co < COUT; co += 8) {
#pragma unroll
    for (int kp = lane; kp < KPAD; kp += 32) {
      f16 v = (f16)0.f;
      if (K == KPAD || kp < K) {  // conv3: folds away (K == KPAD)
        const int rs = kp >> CILOG;       // spatial tap (r*3+s)
        const int ci = kp & (CIN - 1);    // input channel
        v = (f16)wgt[co * K + ci * 9 + rs];
      }
      wlds[co][kp] = v;
    }
  }

  // haloed input patch -> LDS [row][col][ci]
  if constexpr (WIDE) {
    // NHWC: LDS patch rows are straight copies of global rows -> b128 chunks
    constexpr int CW    = CIN / 8;    // v8h chunks per column
    constexpr int CWLOG = clog2(CW);
    constexpr int CPR   = 66 * CW;    // v8h chunks per patch row
#pragma unroll
    for (int rr = 0; rr < 4; ++rr) {
      const int  gy    = y0 + rr;
      const bool rowok = (gy >= 0) && (gy < H);
      const f16* rowp  = (const f16*)in + (long)(b * H + gy) * W * CIN;
      for (int c = tid; c < CPR; c += 256) {
        const int cc  = c >> CWLOG;
        const int ci0 = (c & (CW - 1)) * 8;
        const int gx  = xs + cc - 1;
        v8h v = v8h{};
        if (rowok && gx >= 0 && gx < W) {
          const f16* src = rowp + (long)gx * CIN + ci0;
          v = *(const v8h*)src;
          // warm caches for the next row-pair block (rows y+2..y+5)
          if (gy + 2 < H) __builtin_prefetch(src + 2L * W * CIN, 0, 1);
        }
        *(v8h*)&patch[rr][cc][ci0] = v;
      }
    }
  } else {  // CIN == 1 (conv1 reads fp32 NCHW input)
    const float* inf = (const float*)in + (long)b * H * W;
#pragma unroll
    for (int rr = 0; rr < 4; ++rr) {
      const int  gy    = y0 + rr;
      const bool rowok = (gy >= 0) && (gy < H);
      if (tid < 66) {
        const int gx = xs + tid - 1;
        float v = 0.f;
        if (rowok && gx >= 0 && gx < W) v = inf[(long)gy * W + gx];
        patch[rr][tid][0] = (f16)v;
      }
    }
  }
  __syncthreads();

  for (int p = wave; p < MT * 4; p += 8) {
    const int m   = p % MT;       // M-tile (channels m*16 .. m*16+15)
    const int ct  = p / MT;       // col-tile (cols ct*16 .. +15 of strip)
    const int cc0 = ct * 16 + ln;

    v8f acc[2];
    acc[0] = v8f{0.f, 0.f, 0.f, 0.f, 0.f, 0.f, 0.f, 0.f};
    acc[1] = v8f{0.f, 0.f, 0.f, 0.f, 0.f, 0.f, 0.f, 0.f};

#pragma unroll
    for (int ks = 0; ks < KSTEPS; ++ks) {
      const int k0 = ks * 32;

      // A fragment: contiguous k order matching B
      //   e=0..15 -> k' = k0 + 16*half + e
      const f16* wrow = &wlds[m * 16 + ln][k0 + 16 * half];
      const v8h alo = *(const v8h*)wrow;
      const v8h ahi = *(const v8h*)(wrow + 8);
      const v16h a = __builtin_shufflevector(alo, ahi, 0, 1, 2, 3, 4, 5, 6, 7,
                                             8, 9, 10, 11, 12, 13, 14, 15);

#pragma unroll
      for (int row = 0; row < 2; ++row) {
        // B fragment: k' = k0 + 16*half + e  (e = 0..15)
        v16h bm;
        if constexpr (WIDE) {
          // 16 consecutive k' share one (r,s): rs = idx/CIN, ci0 = idx%CIN
          const int idx = k0 + 16 * half;
          const int rs  = idx >> CILOG;
          const int ci0 = idx & (CIN - 1);
          if (rs < 9) {
            bm = *(const v16h*)&patch[row + rs / 3][cc0 + rs % 3][ci0];
          } else {  // padded tail (conv2: ks==4, half==1)
            bm = v16h{};
          }
        } else {  // CIN == 1: K = 9 -> only half 0, e<9 carry data
          bm = v16h{};
          if (half == 0) {
#pragma unroll
            for (int e = 0; e < 9; ++e)
              bm[e] = patch[row + e / 3][cc0 + e % 3][0];
          }
        }
        acc[row] = __builtin_amdgcn_wmma_f32_16x16x32_f16(
            false, a, false, bm, (short)0, acc[row], false, false);
      }
    }

    const int  colg  = xs + ct * 16 + ln;
    const bool colok = colg < W;

    if (!PASS2) {
      // per-channel sum / sumsq (mask cols beyond W)
#pragma unroll
      for (int v = 0; v < 8; ++v) {
        const int   co = m * 16 + 8 * half + v;
        const float bz = bias[co];
        const float x0 = acc[0][v] + bz, x1 = acc[1][v] + bz;
        float s1 = colok ? (x0 + x1) : 0.f;
        float s2 = colok ? (x0 * x0 + x1 * x1) : 0.f;
        s1 = wred16(s1);
        s2 = wred16(s2);
        if (ln == 0) {  // lanes 0 and 16 (different channels)
          atomicAdd(&sred[co], s1);
          atomicAdd(&sred[COUT + co], s2);
        }
      }
    } else {
      // BN + ReLU + 2x2 maxpool (rows in-wave, cols via shfl_xor(1));
      // lane's 8 values are CONSECUTIVE channels -> one b128 NHWC store
      v8h pk;
#pragma unroll
      for (int v = 0; v < 8; ++v) {
        const int   co = m * 16 + 8 * half + v;
        const float sc = scale[co], sh = shift[co], bz = bias[co];
        const float x0 = fmaxf(sc * (acc[0][v] + bz) + sh, 0.f);
        const float x1 = fmaxf(sc * (acc[1][v] + bz) + sh, 0.f);
        float pm = fmaxf(x0, x1);
        const float nb = __shfl_xor(pm, 1, 32);
        pm = fmaxf(pm, nb);
        pk[v] = (f16)pm;
        if (FUSE_GAP && ((ln & 1) == 0) && colok)
          atomicAdd(&sred[co], pm);  // stage global-avg-pool sum in LDS
      }
      if (!FUSE_GAP && ((ln & 1) == 0) && colok) {
        const int oc = colg >> 1;
        f16* dst = outp + (((long)b * (H / 2) + yp) * (W / 2) + oc) * COUT +
                   m * 16 + 8 * half;
        *(v8h*)dst = pk;
      }
    }
  }
  __syncthreads();
  if (!PASS2) {
    if (tid < 2 * COUT) atomicAdd(&stats[tid], sred[tid]);
  } else if (FUSE_GAP) {
    if (tid < COUT) atomicAdd(&gap[(long)b * COUT + tid], sred[tid]);
  }
}

// ---------------------------------------------------------------------------
__global__ void bn_finalize(const float* __restrict__ stats,
                            const float* __restrict__ g,
                            const float* __restrict__ bb,
                            float* __restrict__ scale, float* __restrict__ shift,
                            int C, float inv_cnt) {
  const int c = threadIdx.x;
  if (c < C) {
    const float mean = stats[c] * inv_cnt;
    const float var  = stats[C + c] * inv_cnt - mean * mean;
    const float sc   = g[c] * rsqrtf(var + 1e-5f);
    scale[c] = sc;
    shift[c] = bb[c] - mean * sc;
  }
}

__global__ void zero_ws(float* __restrict__ ws) {
  const int i = blockIdx.x * 256 + threadIdx.x;
  if (i < 384) ws[i] = 0.f;            // stats for 3 BN stages
  if (i < 16384) ws[1024 + i] = 0.f;   // gap accumulator (256x64)
}

// ---------------------------------------------------------------------------
// Head: one wave32 per batch element.
//   gap/784 -> fc1(64->5) -> LayerNorm(5) -> 5-qubit/3-layer circuit
//   (lane == basis state, gates = shfl butterflies/permutes)
//   -> fc2(1->32)+ReLU -> fc3(32->2) -> log_softmax
// ---------------------------------------------------------------------------
__global__ __launch_bounds__(256) void head_kernel(
    const float* __restrict__ gap, const float* __restrict__ fc1w,
    const float* __restrict__ fc1b, const float* __restrict__ lng,
    const float* __restrict__ lnb, const float* __restrict__ qp,
    const float* __restrict__ fc2w, const float* __restrict__ fc2b,
    const float* __restrict__ fc3w, const float* __restrict__ fc3b,
    float* __restrict__ out) {
  const int lane = threadIdx.x & 31;
  const int b = blockIdx.x * 8 + (threadIdx.x >> 5);

  // fc1: h[j] = sum_c gap[b][c]/784 * W[c][j] + b[j]
  const float inv = 1.0f / 784.0f;
  const float g0 = gap[b * 64 + lane] * inv;
  const float g1 = gap[b * 64 + 32 + lane] * inv;
  float h[5];
#pragma unroll
  for (int j = 0; j < 5; ++j) {
    float v = g0 * fc1w[lane * 5 + j] + g1 * fc1w[(lane + 32) * 5 + j];
    h[j] = wred32(v) + fc1b[j];
  }
  // LayerNorm over 5
  float mu = 0.f;
#pragma unroll
  for (int j = 0; j < 5; ++j) mu += h[j];
  mu *= 0.2f;
  float var = 0.f;
#pragma unroll
  for (int j = 0; j < 5; ++j) { const float d = h[j] - mu; var += d * d; }
  var *= 0.2f;
  const float rs = rsqrtf(var + 1e-5f);
#pragma unroll
  for (int j = 0; j < 5; ++j) h[j] = lng[j] * (h[j] - mu) * rs + lnb[j];

  // quantum circuit: lane = basis index; wire w -> bit (4-w)
  float ar = (lane == 0) ? 1.f : 0.f, ai = 0.f;
#pragma unroll
  for (int l = 0; l < 3; ++l) {
#pragma unroll
    for (int i = 0; i < 5; ++i) {  // RX(h[i]) on wire i
      const int msk = 1 << (4 - i);
      const float pr = __shfl_xor(ar, msk, 32);
      const float pi = __shfl_xor(ai, msk, 32);
      const float c = cosf(h[i] * 0.5f), s = sinf(h[i] * 0.5f);
      const float nr = c * ar + s * pi;
      const float ni = c * ai - s * pr;
      ar = nr; ai = ni;
    }
#pragma unroll
    for (int i = 0; i < 5; ++i) {  // RY then RZ on wire i
      const float ty = qp[(l * 10 + i) % 30];
      const float tz = qp[(l * 10 + i + 5) % 30];
      const int msk = 1 << (4 - i);
      const int bit = (lane >> (4 - i)) & 1;
      // RY
      {
        const float pr = __shfl_xor(ar, msk, 32);
        const float pi = __shfl_xor(ai, msk, 32);
        const float c = cosf(ty * 0.5f), s = sinf(ty * 0.5f);
        const float sg = bit ? s : -s;
        ar = c * ar + sg * pr;
        ai = c * ai + sg * pi;
      }
      // RZ (diagonal phase)
      {
        const float cr = cosf(tz * 0.5f);
        const float si = bit ? sinf(tz * 0.5f) : -sinf(tz * 0.5f);
        const float nr = ar * cr - ai * si;
        const float ni = ar * si + ai * cr;
        ar = nr; ai = ni;
      }
    }
#pragma unroll
    for (int e = 0; e < 5; ++e) {  // CNOT ring (0,1)(1,2)(2,3)(3,4)(4,0)
      const int cw = e, tw = (e + 1) % 5;
      const int cb = (lane >> (4 - cw)) & 1;
      const int src = cb ? (lane ^ (1 << (4 - tw))) : lane;
      ar = __shfl(ar, src, 32);
      ai = __shfl(ai, src, 32);
    }
  }
  // <Z0 Z1 Z2>
  const float prob = ar * ar + ai * ai;
  const int bits = (lane >> 2) & 7;
  const float sg = (__popc(bits) & 1) ? -1.f : 1.f;
  const float q = wred32(prob * sg);

  // fc2(1->32)+ReLU ; fc3(32->2) ; log_softmax
  const float z = fmaxf(q * fc2w[lane] + fc2b[lane], 0.f);
  const float t0 = wred32(z * fc3w[lane * 2 + 0]) + fc3b[0];
  const float t1 = wred32(z * fc3w[lane * 2 + 1]) + fc3b[1];
  const float mx = fmaxf(t0, t1);
  const float lse = mx + logf(expf(t0 - mx) + expf(t1 - mx));
  if (lane == 0) {
    out[b * 2 + 0] = t0 - lse;
    out[b * 2 + 1] = t1 - lse;
  }
}

// ---------------------------------------------------------------------------
extern "C" void kernel_launch(void* const* d_in, const int* in_sizes, int n_in,
                              void* d_out, int out_size, void* d_ws,
                              size_t ws_size, hipStream_t stream) {
  const float* x       = (const float*)d_in[0];
  const float* conv1_w = (const float*)d_in[1];
  const float* conv1_b = (const float*)d_in[2];
  const float* bn1_g   = (const float*)d_in[3];
  const float* bn1_b   = (const float*)d_in[4];
  const float* conv2_w = (const float*)d_in[5];
  const float* conv2_b = (const float*)d_in[6];
  const float* bn2_g   = (const float*)d_in[7];
  const float* bn2_b   = (const float*)d_in[8];
  const float* conv3_w = (const float*)d_in[9];
  const float* conv3_b = (const float*)d_in[10];
  const float* bn3_g   = (const float*)d_in[11];
  const float* bn3_b   = (const float*)d_in[12];
  const float* fc1_w   = (const float*)d_in[13];
  const float* fc1_b   = (const float*)d_in[14];
  const float* ln_g    = (const float*)d_in[15];
  const float* ln_b    = (const float*)d_in[16];
  const float* q_par   = (const float*)d_in[17];
  const float* fc2_w   = (const float*)d_in[18];
  const float* fc2_b   = (const float*)d_in[19];
  const float* fc3_w   = (const float*)d_in[20];
  const float* fc3_b   = (const float*)d_in[21];

  // workspace layout (floats at base; f16 NHWC activations after 128KB)
  float* wsf    = (float*)d_ws;
  float* stats1 = wsf + 0;            // [2][64] used [2][16]
  float* stats2 = wsf + 128;
  float* stats3 = wsf + 256;
  float* sc1 = wsf + 384, *sh1 = wsf + 448;
  float* sc2 = wsf + 512, *sh2 = wsf + 576;
  float* sc3 = wsf + 640, *sh3 = wsf + 704;
  float* gap = wsf + 1024;            // [256][64]
  f16* h1 = (f16*)((char*)d_ws + 131072);                  // 256x112x112x16
  f16* h2 = (f16*)((char*)d_ws + 131072 + 102760448UL);    // 256x56x56x32

  zero_ws<<<64, 256, 0, stream>>>(wsf);

  // ---- stage 1: 1 -> 16 @ 224x224 ----
  conv_wmma<float, 1, 16, 224, 224, false, false>
      <<<256 * 112 * 4, 256, 0, stream>>>(x, conv1_w, conv1_b, stats1,
                                          nullptr, nullptr, nullptr, nullptr);
  bn_finalize<<<1, 64, 0, stream>>>(stats1, bn1_g, bn1_b, sc1, sh1, 16,
                                    1.0f / (256.0f * 224.0f * 224.0f));
  conv_wmma<float, 1, 16, 224, 224, true, false>
      <<<256 * 112 * 4, 256, 0, stream>>>(x, conv1_w, conv1_b, nullptr,
                                          sc1, sh1, h1, nullptr);

  // ---- stage 2: 16 -> 32 @ 112x112 ----
  conv_wmma<f16, 16, 32, 112, 112, false, false>
      <<<256 * 56 * 2, 256, 0, stream>>>(h1, conv2_w, conv2_b, stats2,
                                         nullptr, nullptr, nullptr, nullptr);
  bn_finalize<<<1, 64, 0, stream>>>(stats2, bn2_g, bn2_b, sc2, sh2, 32,
                                    1.0f / (256.0f * 112.0f * 112.0f));
  conv_wmma<f16, 16, 32, 112, 112, true, false>
      <<<256 * 56 * 2, 256, 0, stream>>>(h1, conv2_w, conv2_b, nullptr,
                                         sc2, sh2, h2, nullptr);

  // ---- stage 3: 32 -> 64 @ 56x56 (fused global-avg-pool) ----
  conv_wmma<f16, 32, 64, 56, 56, false, false>
      <<<256 * 28 * 1, 256, 0, stream>>>(h2, conv3_w, conv3_b, stats3,
                                         nullptr, nullptr, nullptr, nullptr);
  bn_finalize<<<1, 64, 0, stream>>>(stats3, bn3_g, bn3_b, sc3, sh3, 64,
                                    1.0f / (256.0f * 56.0f * 56.0f));
  conv_wmma<f16, 32, 64, 56, 56, true, true>
      <<<256 * 28 * 1, 256, 0, stream>>>(h2, conv3_w, conv3_b, nullptr,
                                         sc3, sh3, nullptr, gap);

  // ---- head: fc1 + LN + quantum circuit + fc2/fc3 + log_softmax ----
  head_kernel<<<32, 256, 0, stream>>>(gap, fc1_w, fc1_b, ln_g, ln_b, q_par,
                                      fc2_w, fc2_b, fc3_w, fc3_b,
                                      (float*)d_out);
}
